// my_model_84834194031115
// MI455X (gfx1250) — matrix operand order repaired and checked by reading, not compiled, not run on hardware
//
#include <hip/hip_runtime.h>
#include <math.h>

typedef __attribute__((ext_vector_type(16))) _Float16 v16h;
typedef __attribute__((ext_vector_type(8)))  _Float16 v8h;
typedef __attribute__((ext_vector_type(8)))  float    v8f;

#define B_ 8
#define T_ 128
#define E_ 256
#define H_ 512
#define G_ 2048   // 4*H

__device__ __forceinline__ float sigf(float x) { return 1.0f / (1.0f + expf(-x)); }

// ---------------------------------------------------------------------------
// Embedding gather, converted straight to f16 (feeds only the WMMA GEMMs).
// grid = B*T rows, block = E threads.
// ---------------------------------------------------------------------------
__global__ void embed_gather_f16(const int* __restrict__ tokens,
                                 const float* __restrict__ table,
                                 _Float16* __restrict__ embh) {
  const int row = blockIdx.x;            // b*T + t
  const int e   = threadIdx.x;           // 0..E-1
  const int tok = tokens[row];
  embh[(size_t)row * E_ + e] = (_Float16)table[(size_t)tok * E_ + e];
}

// ---------------------------------------------------------------------------
// Convert f32 (K x N) -> f16 TRANSPOSED (N x K), LDS-tiled so both the global
// read and the global write are coalesced. Grid: (K/32, N/32), 256 threads.
// Transposed weights make every WMMA B-fragment a contiguous, 32B-aligned
// 16-half run -> two global_load_b128 instead of 16 global_load_u16.
// ---------------------------------------------------------------------------
__global__ void f32_to_f16_T(const float* __restrict__ src,  // K x N
                             _Float16* __restrict__ dst,     // N x K
                             int K, int N) {
  __shared__ float tile[32][33];
  const int kb = blockIdx.x * 32;
  const int nb = blockIdx.y * 32;
  const int tx = threadIdx.x & 31;
  const int ty = threadIdx.x >> 5;       // 0..7
#pragma unroll
  for (int r = ty; r < 32; r += 8)
    tile[r][tx] = src[(size_t)(kb + r) * N + nb + tx];
  __syncthreads();
#pragma unroll
  for (int r = ty; r < 32; r += 8)
    dst[(size_t)(nb + r) * K + kb + tx] = (_Float16)tile[tx][r];
}

// ---------------------------------------------------------------------------
// Y(MxN, f32) = A(MxK, f16) @ B(KxN) + bias(N), with B given TRANSPOSED as
// Bt(N x K, f16). One 16x16 output tile per wave; 8 waves per block.
// A fragment per lane (m = lane&15, khalf = lane>>4):
//   a[0..7]  = A[m][k0 + khalf*16 .. +7]
//   a[8..15] = A[m][k0 + khalf*16 + 8 .. +7]
// B fragment per lane (n = lane&15):
//   b[j] = B[k0 + khalf*16 + j][n] = Bt[n][k0 + khalf*16 + j]  (contiguous!)
// D: row = r + khalf*8, col = lane&15.
// ---------------------------------------------------------------------------
__global__ void gemm_bias_f16(const _Float16* __restrict__ A,   // M x K
                              const _Float16* __restrict__ Bt,  // N x K
                              const float* __restrict__ bias,   // N
                              float* __restrict__ Y,            // M x N
                              int M, int N, int K) {
  const int wid   = threadIdx.x >> 5;
  const int lane  = threadIdx.x & 31;
  const int khalf = lane >> 4;
  const int l16   = lane & 15;
  const int tile  = blockIdx.x * 8 + wid;
  const int ntn   = N >> 4;
  const int mt    = tile / ntn;
  const int nt    = tile - mt * ntn;

  const _Float16* arow = A  + (size_t)(mt * 16 + l16) * K;
  const _Float16* brow = Bt + (size_t)(nt * 16 + l16) * K;

  v8f acc = {};
  for (int k0 = 0; k0 < K; k0 += 32) {
    const _Float16* ap = arow + k0 + khalf * 16;
    v8h lo = *(const v8h*)ap;
    v8h hi = *(const v8h*)(ap + 8);
    v16h a = __builtin_shufflevector(lo, hi, 0, 1, 2, 3, 4, 5, 6, 7,
                                             8, 9, 10, 11, 12, 13, 14, 15);
    v16h b = *(const v16h*)(brow + k0 + khalf * 16);   // 32B aligned
    acc = __builtin_amdgcn_wmma_f32_16x16x32_f16(false, a, false, b,
                                                 (short)0, acc, false, false);
  }
  const int   col = nt * 16 + l16;
  const float bv  = bias[col];
#pragma unroll
  for (int r = 0; r < 8; ++r) {
    const int row = mt * 16 + khalf * 8 + r;
    Y[(size_t)row * N + col] = acc[r] + bv;
  }
}

// ---------------------------------------------------------------------------
// cadd(B x G) = h_last(B x H) @ Wmh(H x G)   (tiny, f32, coalesced on n)
// ---------------------------------------------------------------------------
__global__ void matvec_hW(const float* __restrict__ hl,
                          const float* __restrict__ W,
                          float* __restrict__ cadd) {
  const int idx = blockIdx.x * 256 + threadIdx.x;   // B*G = 16384
  const int bb  = idx >> 11;
  const int n   = idx & (G_ - 1);
  float s = 0.0f;
  for (int k = 0; k < H_; ++k) s += hl[bb * H_ + k] * W[(size_t)k * G_ + n];
  cadd[idx] = s;
}

// ---------------------------------------------------------------------------
// Persistent single-workgroup LSTM scan (1024 threads = 32 waves, one WGP).
// Per step: pre = h16 @ Whh (WMMA, M=16 padded from B=8, N=2048, K=512) with
// WhhT given transposed (G x H, f16), then gates in f32 from LDS.
// Loop nest is k0 -> N-tile so each LDS A-fragment is loaded ONCE per k-step
// and reused by 4 back-to-back WMMAs (4 live accumulators).
// X holds the precomputed input projection (+bias); cadd is the optional
// constant R_t @ Wmh_t term (task LSTM, provably == h_last @ Wmh_t).
// write_all=1: emit h each step to out (B,T,H); else only final h (B,H).
// ---------------------------------------------------------------------------
__global__ void lstm_scan(const float* __restrict__ X,       // (B*T) x G
                          const _Float16* __restrict__ WhhT, // G x H, f16 (transposed)
                          const float* __restrict__ cadd,    // B x G or null
                          int use_cadd,
                          float* __restrict__ out,
                          int write_all) {
  extern __shared__ char smem[];
  float*    preS = (float*)smem;                 // B*G   f32 = 64 KB
  float*    cS   = preS + B_ * G_;               // B*H   f32 = 16 KB
  _Float16* h16  = (_Float16*)(cS + B_ * H_);    // 16*H  f16 = 16 KB (rows 8..15 = 0 pad)

  const int tid   = threadIdx.x;                 // 0..1023
  const int wid   = tid >> 5;
  const int lane  = tid & 31;
  const int khalf = lane >> 4;
  const int l16   = lane & 15;

  for (int i = tid; i < B_ * H_; i += 1024) cS[i] = 0.0f;
  for (int i = tid; i < 16 * H_; i += 1024) h16[i] = (_Float16)0.0f;
  __syncthreads();

  const _Float16* hrow = h16 + l16 * H_;
  // 4 N-tiles per wave: nt = wid + tt*32; hoist the Bt row pointers.
  const _Float16* brow0 = WhhT + (size_t)((wid +  0) * 16 + l16) * H_;
  const _Float16* brow1 = WhhT + (size_t)((wid + 32) * 16 + l16) * H_;
  const _Float16* brow2 = WhhT + (size_t)((wid + 64) * 16 + l16) * H_;
  const _Float16* brow3 = WhhT + (size_t)((wid + 96) * 16 + l16) * H_;

  for (int t = 0; t < T_; ++t) {
    // ---- recurrent GEMM: k-outer, 4 N-tiles inner (A loaded once per k) ----
    v8f a0 = {}, a1 = {}, a2 = {}, a3 = {};
    for (int k0 = 0; k0 < H_; k0 += 32) {
      const _Float16* hp = hrow + k0 + khalf * 16;
      v8h lo = *(const v8h*)hp;            // ds_load_b128
      v8h hi = *(const v8h*)(hp + 8);
      v16h a = __builtin_shufflevector(lo, hi, 0, 1, 2, 3, 4, 5, 6, 7,
                                               8, 9, 10, 11, 12, 13, 14, 15);
      const int ko = k0 + khalf * 16;
      v16h b0 = *(const v16h*)(brow0 + ko);  // 2x global_load_b128 each
      v16h b1 = *(const v16h*)(brow1 + ko);
      v16h b2 = *(const v16h*)(brow2 + ko);
      v16h b3 = *(const v16h*)(brow3 + ko);
      a0 = __builtin_amdgcn_wmma_f32_16x16x32_f16(false, a, false, b0, (short)0, a0, false, false);
      a1 = __builtin_amdgcn_wmma_f32_16x16x32_f16(false, a, false, b1, (short)0, a1, false, false);
      a2 = __builtin_amdgcn_wmma_f32_16x16x32_f16(false, a, false, b2, (short)0, a2, false, false);
      a3 = __builtin_amdgcn_wmma_f32_16x16x32_f16(false, a, false, b3, (short)0, a3, false, false);
    }
    // rows 0..7 of each D tile are the real batch rows (lanes 0..15, khalf==0)
    if (lane < 16) {
#pragma unroll
      for (int r = 0; r < 8; ++r) {
        preS[r * G_ + (wid +  0) * 16 + l16] = a0[r];
        preS[r * G_ + (wid + 32) * 16 + l16] = a1[r];
        preS[r * G_ + (wid + 64) * 16 + l16] = a2[r];
        preS[r * G_ + (wid + 96) * 16 + l16] = a3[r];
      }
    }
    __syncthreads();

    // ---- gates (f32): 8*512 elements over 1024 threads ----
    for (int e = tid; e < B_ * H_; e += 1024) {
      const int bb = e >> 9;
      const int n  = e & (H_ - 1);
      const float* xr = X + (size_t)(bb * T_ + t) * G_;
      const float* pr = preS + bb * G_;
      float pi = pr[n]          + xr[n];
      float pf = pr[H_ + n]     + xr[H_ + n];
      float pg = pr[2 * H_ + n] + xr[2 * H_ + n];
      float po = pr[3 * H_ + n] + xr[3 * H_ + n];
      if (use_cadd) {
        const float* cr = cadd + bb * G_;
        pi += cr[n]; pf += cr[H_ + n]; pg += cr[2 * H_ + n]; po += cr[3 * H_ + n];
      }
      const float iv = sigf(pi);
      const float fv = sigf(pf);
      const float ov = sigf(po);
      const float gv = tanhf(pg);
      const float cn = fv * cS[e] + iv * gv;
      const float hn = ov * tanhf(cn);
      cS[e] = cn;
      h16[bb * H_ + n] = (_Float16)hn;
      if (write_all)        out[(size_t)(bb * T_ + t) * H_ + n] = hn;
      else if (t == T_ - 1) out[(size_t)bb * H_ + n]            = hn;
    }
    __syncthreads();
  }
}

// ---------------------------------------------------------------------------
// Host-side orchestration
// ---------------------------------------------------------------------------
extern "C" void kernel_launch(void* const* d_in, const int* in_sizes, int n_in,
                              void* d_out, int out_size, void* d_ws, size_t ws_size,
                              hipStream_t stream) {
  (void)in_sizes; (void)n_in; (void)out_size; (void)ws_size;

  const int*   tokens = (const int*)  d_in[0];
  const float* table  = (const float*)d_in[2];
  const float* Wih_s  = (const float*)d_in[3];
  const float* Whh_s  = (const float*)d_in[4];
  const float* b_s    = (const float*)d_in[5];
  // d_in[6..9] (Ws_w, Ws_b, Us_w, Us_b) are provably dead: softmax weights sum
  // to 1 and h_shared is t-invariant, so Rt == h_last exactly.
  const float* Wih_t  = (const float*)d_in[10];
  const float* Whh_t  = (const float*)d_in[11];
  const float* Wmh_t  = (const float*)d_in[12];
  const float* b_t    = (const float*)d_in[13];

  char*  ws  = (char*)d_ws;
  size_t off = 0;
  auto alloc = [&](size_t bytes) -> void* {
    void* p = ws + off;
    off += (bytes + 255) & ~(size_t)255;
    return p;
  };

  _Float16* embh  = (_Float16*)alloc((size_t)B_ * T_ * E_ * 2);
  _Float16* WihSt = (_Float16*)alloc((size_t)E_ * G_ * 2);   // G x E (transposed)
  _Float16* WihTt = (_Float16*)alloc((size_t)E_ * G_ * 2);   // G x E
  _Float16* WhhSt = (_Float16*)alloc((size_t)H_ * G_ * 2);   // G x H
  _Float16* WhhTt = (_Float16*)alloc((size_t)H_ * G_ * 2);   // G x H
  float*    Xs    = (float*)   alloc((size_t)B_ * T_ * G_ * 4);
  float*    Xt    = (float*)   alloc((size_t)B_ * T_ * G_ * 4);
  float*    hlast = (float*)   alloc((size_t)B_ * H_ * 4);
  float*    cadd  = (float*)   alloc((size_t)B_ * G_ * 4);

  // 1) transposed f16 weight conversions + embedding gather (all independent)
  {
    dim3 gE(E_ / 32, G_ / 32), gH(H_ / 32, G_ / 32);
    f32_to_f16_T<<<gE, 256, 0, stream>>>(Wih_s, WihSt, E_, G_);
    f32_to_f16_T<<<gE, 256, 0, stream>>>(Wih_t, WihTt, E_, G_);
    f32_to_f16_T<<<gH, 256, 0, stream>>>(Whh_s, WhhSt, H_, G_);
    f32_to_f16_T<<<gH, 256, 0, stream>>>(Whh_t, WhhTt, H_, G_);
  }
  embed_gather_f16<<<B_ * T_, E_, 0, stream>>>(tokens, table, embh);

  // 2) parallel input projections: X = emb @ Wih + bias  (WMMA)
  const int M     = B_ * T_;                 // 1024
  const int tiles = (M / 16) * (G_ / 16);    // 8192 tiles, 8 per block
  gemm_bias_f16<<<tiles / 8, 256, 0, stream>>>(embh, WihSt, b_s, Xs, M, G_, E_);
  gemm_bias_f16<<<tiles / 8, 256, 0, stream>>>(embh, WihTt, b_t, Xt, M, G_, E_);

  // 3) sequential scans (persistent single-WGP kernels)
  const size_t lds = (size_t)B_ * G_ * 4 + (size_t)B_ * H_ * 4 + (size_t)16 * H_ * 2; // 96 KB
  lstm_scan<<<1, 1024, lds, stream>>>(Xs, WhhSt, nullptr, 0, hlast, 0);
  matvec_hW<<<(B_ * G_) / 256, 256, 0, stream>>>(hlast, Wmh_t, cadd);
  lstm_scan<<<1, 1024, lds, stream>>>(Xt, WhhTt, cadd, 1, (float*)d_out, 1);
}
